// ParallelMemoryModule_10995116277898
// MI455X (gfx1250) — matrix operand order repaired and checked by reading, not compiled, not run on hardware
//
#include <hip/hip_runtime.h>
#include <cstdint>
#include <cstddef>

// ---------------------------------------------------------------------------
// Problem constants
// ---------------------------------------------------------------------------
#define S_  2048
#define B_  2
#define H_  2048
#define NH_ 16
#define D_  128
#define M_  2048
#define FF_ 8192
#define ROWS_ (S_ * B_)   // 4096
#define MBROWS_ (M_ * B_) // 4096
#define INV_NORM_ 0.08838834764831845f  // 1/sqrt(128)

typedef __attribute__((ext_vector_type(16))) __bf16 v16bf;
typedef __attribute__((ext_vector_type(8)))  float  v8f;

union FragU { uint4 u[2]; v16bf v; };

__device__ __forceinline__ unsigned short f2bf(float f) {
    unsigned int u = __float_as_uint(f);
    unsigned int r = u + 0x7FFFu + ((u >> 16) & 1u);   // round-to-nearest-even
    return (unsigned short)(r >> 16);
}

__device__ __forceinline__ float gelu_f(float x) {
    float x3 = x * x * x;
    float t = tanhf(0.7978845608028654f * (x + 0.044715f * x3));
    return 0.5f * x * (1.0f + t);
}

// ---------------------------------------------------------------------------
// CDNA5 async global->LDS copy (VGLOBAL GLOBAL_LOAD_ASYNC_TO_LDS_B128).
// Per-lane: LDS[lds_off .. +15] = MEM[gaddr .. +15]; tracked by ASYNCcnt.
// ---------------------------------------------------------------------------
__device__ __forceinline__ void async_copy_b128(void* lds_ptr, const void* gaddr) {
    unsigned lds_off = (unsigned)(unsigned long long)lds_ptr;   // addrspace(3) offset
    asm volatile("global_load_async_to_lds_b128 %0, %1, off"
                 :: "v"(lds_off), "v"((unsigned long long)gaddr)
                 : "memory");
}

__device__ __forceinline__ void wait_async0() {
#if __has_builtin(__builtin_amdgcn_s_wait_asynccnt)
    __builtin_amdgcn_s_wait_asynccnt(0);
#else
    asm volatile("s_wait_asynccnt 0x0" ::: "memory");
#endif
}

// ---------------------------------------------------------------------------
// f32 -> bf16 conversion (grid-stride)
// ---------------------------------------------------------------------------
__global__ void cvt_bf16_kernel(const float* __restrict__ in,
                                unsigned short* __restrict__ out, size_t n) {
    size_t i = (size_t)blockIdx.x * blockDim.x + threadIdx.x;
    size_t stride = (size_t)gridDim.x * blockDim.x;
    for (; i < n; i += stride) out[i] = f2bf(in[i]);
}

// ---------------------------------------------------------------------------
// f32 [K][N] -> bf16 transposed [N][K] (32x32 tiles via LDS, coalesced)
// ---------------------------------------------------------------------------
__global__ __launch_bounds__(256) void cvt_transpose_kernel(
    const float* __restrict__ in, unsigned short* __restrict__ out,
    int K, int N) {
    __shared__ float tile[32][33];
    const int kt = blockIdx.y * 32;
    const int nt = blockIdx.x * 32;
    const int tx = threadIdx.x & 31;
    const int ty = threadIdx.x >> 5;          // 0..7
#pragma unroll
    for (int r = ty; r < 32; r += 8)
        tile[r][tx] = in[(size_t)(kt + r) * N + nt + tx];
    __syncthreads();
#pragma unroll
    for (int r = ty; r < 32; r += 8)          // out[n][k], k contiguous
        out[(size_t)(nt + r) * K + kt + tx] = f2bf(tile[tx][r]);
}

// ---------------------------------------------------------------------------
// LayerNorm over last dim H (one block per row), f32 in -> bf16 out
// ---------------------------------------------------------------------------
__global__ __launch_bounds__(256) void layernorm_kernel(
    const float* __restrict__ x, const float* __restrict__ g,
    const float* __restrict__ b, unsigned short* __restrict__ out, int H) {
    int row = blockIdx.x;
    const float* xr = x + (size_t)row * H;
    float s = 0.f, ss = 0.f;
    for (int i = threadIdx.x; i < H; i += 256) {
        float v = xr[i]; s += v; ss += v * v;
    }
    __shared__ float rs[256], rss[256];
    rs[threadIdx.x] = s; rss[threadIdx.x] = ss;
    __syncthreads();
    for (int o = 128; o > 0; o >>= 1) {
        if (threadIdx.x < (unsigned)o) {
            rs[threadIdx.x]  += rs[threadIdx.x + o];
            rss[threadIdx.x] += rss[threadIdx.x + o];
        }
        __syncthreads();
    }
    float mu  = rs[0] / (float)H;
    float var = rss[0] / (float)H - mu * mu;
    float inv = rsqrtf(var + 1e-5f);
    for (int i = threadIdx.x; i < H; i += 256)
        out[(size_t)row * H + i] = f2bf((xr[i] - mu) * inv * g[i] + b[i]);
}

// ---------------------------------------------------------------------------
// Tiled bf16 WMMA GEMM:  C[Mr,N] = A[Mr,K] @ W^T   (W supplied as [N][K])
// (+bias, opt gelu/resid). Block tile 128x128x32, 256 threads = 8 waves
// (2x4), wave tile 64x32 (4x2 fragments). Double-buffered LDS filled by
// GLOBAL_LOAD_ASYNC_TO_LDS_B128 (no staging registers, no spills).
// ---------------------------------------------------------------------------
__global__ __launch_bounds__(256)
void gemm_bf16_kernel(
    const unsigned short* __restrict__ A,   // [Mr][K]
    const unsigned short* __restrict__ Wt,  // [N][K]  (pre-transposed weight)
    const float* __restrict__ bias,
    const float* __restrict__ resid,
    float* __restrict__ outF,
    unsigned short* __restrict__ outB,
    int Mr, int N, int K, int act_gelu) {

    __shared__ __align__(16) unsigned short As[2][128 * 40]; // [m][k] pad 8
    __shared__ __align__(16) unsigned short Bt[2][128 * 40]; // [n][k] pad 8

    const int tid  = threadIdx.x;
    const int wid  = tid >> 5;
    const int lane = tid & 31;
    const int half = lane >> 4;
    const int l    = lane & 15;
    const int m0 = blockIdx.y * 128;
    const int n0 = blockIdx.x * 128;
    const int wm = (wid & 1) * 64;   // wave row offset inside tile (2 groups)
    const int wn = (wid >> 1) * 32;  // wave col offset inside tile (4 groups)

    // staging coordinates: 512 b128 chunks each for A and B, 2 per thread
    int crow[2], ckb[2];
#pragma unroll
    for (int t = 0; t < 2; ++t) {
        int c = tid + t * 256;
        crow[t] = c >> 2;            // 0..127
        ckb[t]  = (c & 3) * 8;       // 0..24
    }

    v8f acc[4][2];
#pragma unroll
    for (int i = 0; i < 4; ++i)
#pragma unroll
        for (int j = 0; j < 2; ++j) acc[i][j] = (v8f)0.0f;

    // prologue: async-stage first tile into buffer 0
#pragma unroll
    for (int t = 0; t < 2; ++t) {
        async_copy_b128(&As[0][crow[t] * 40 + ckb[t]],
                        A + (size_t)(m0 + crow[t]) * K + ckb[t]);
        async_copy_b128(&Bt[0][crow[t] * 40 + ckb[t]],
                        Wt + (size_t)(n0 + crow[t]) * K + ckb[t]);
    }

    int buf = 0;
    for (int k0 = 0; k0 < K; k0 += 32) {
        wait_async0();        // my async copies into current buffer complete
        __syncthreads();      // everyone's complete; prior reads of nb done
        const bool haveNext = (k0 + 32 < K);
        if (haveNext) {
            int kn = k0 + 32, nb = buf ^ 1;
#pragma unroll
            for (int t = 0; t < 2; ++t) {
                async_copy_b128(&As[nb][crow[t] * 40 + ckb[t]],
                                A + (size_t)(m0 + crow[t]) * K + kn + ckb[t]);
                async_copy_b128(&Bt[nb][crow[t] * 40 + ckb[t]],
                                Wt + (size_t)(n0 + crow[t]) * K + kn + ckb[t]);
            }
        }

        // A fragments: lane l -> row M=l; halves take K {0..7,16..23}/{8..15,24..31}
        v16bf af[4];
#pragma unroll
        for (int mf = 0; mf < 4; ++mf) {
            int row = wm + mf * 16 + l;
            int ko  = half * 8;
            FragU tmp;
            tmp.u[0] = *(const uint4*)(&As[buf][row * 40 + ko]);
            tmp.u[1] = *(const uint4*)(&As[buf][row * 40 + ko + 16]);
            af[mf] = tmp.v;
        }
#pragma unroll
        for (int nf = 0; nf < 2; ++nf) {
            int col = wn + nf * 16 + l;
            int ko  = half * 8;                  // B: halves take K {0..7,16..23}/{8..15,24..31}
            FragU tb;
            tb.u[0] = *(const uint4*)(&Bt[buf][col * 40 + ko]);
            tb.u[1] = *(const uint4*)(&Bt[buf][col * 40 + ko + 16]);
            v16bf bf = tb.v;
#pragma unroll
            for (int mf = 0; mf < 4; ++mf)
                acc[mf][nf] = __builtin_amdgcn_wmma_f32_16x16x32_bf16(
                    false, af[mf], false, bf, (short)0, acc[mf][nf], false, false);
        }
        buf ^= 1;
    }

    // epilogue: C frag element: VGPR e -> row e+8*half, col l
#pragma unroll
    for (int mf = 0; mf < 4; ++mf) {
#pragma unroll
        for (int nf = 0; nf < 2; ++nf) {
            int gcol = n0 + wn + nf * 16 + l;
            float bia = bias[gcol];
#pragma unroll
            for (int e = 0; e < 8; ++e) {
                int grow = m0 + wm + mf * 16 + e + 8 * half;
                float v = acc[mf][nf][e] + bia;
                if (act_gelu) v = gelu_f(v);
                if (resid) v += resid[(size_t)grow * N + gcol];
                if (outF) outF[(size_t)grow * N + gcol] = v;
                if (outB) outB[(size_t)grow * N + gcol] = f2bf(v);
            }
        }
    }
}

// ---------------------------------------------------------------------------
// L2 normalize q heads: qraw f32 [(s*B+b)*H + h*D + d] -> qb bf16 [b][h][s][d]
// ---------------------------------------------------------------------------
__global__ __launch_bounds__(128) void l2norm_q_kernel(
    const float* __restrict__ qraw, unsigned short* __restrict__ qout) {
    int bi = blockIdx.x;                 // s * (B*NH) + b*NH + h
    int s = bi / (B_ * NH_);
    int rem = bi % (B_ * NH_);
    int b = rem / NH_, h = rem % NH_;
    int d = threadIdx.x;
    float v = qraw[((size_t)s * B_ + b) * H_ + h * D_ + d];
    __shared__ float red[128];
    red[d] = v * v;
    __syncthreads();
    for (int o = 64; o > 0; o >>= 1) {
        if (d < o) red[d] += red[d + o];
        __syncthreads();
    }
    float inv = rsqrtf(red[0] + 1e-12f);
    qout[(((size_t)b * NH_ + h) * S_ + s) * D_ + d] = f2bf(v * inv);
}

// ---------------------------------------------------------------------------
// L2 normalize k,v heads from kvraw f32 [(m*B+b)*2H + h*2D + {d | D+d}].
//   k -> bf16 [b][h][m][d]   (natural, B-operand of QK^T)
//   v -> bf16 [b][h][d][m]   (pre-transposed, B-operand of P*V)
// ---------------------------------------------------------------------------
__global__ __launch_bounds__(128) void l2norm_kv_kernel(
    const float* __restrict__ kvraw, unsigned short* __restrict__ kout,
    unsigned short* __restrict__ vtout) {
    int bi = blockIdx.x;
    int m = bi / (B_ * NH_);
    int rem = bi % (B_ * NH_);
    int b = rem / NH_, h = rem % NH_;
    int d = threadIdx.x;
    size_t base = ((size_t)m * B_ + b) * (2 * H_) + h * (2 * D_);
    float kv = kvraw[base + d];
    float vv = kvraw[base + D_ + d];
    __shared__ float rk[128], rv[128];
    rk[d] = kv * kv; rv[d] = vv * vv;
    __syncthreads();
    for (int o = 64; o > 0; o >>= 1) {
        if (d < o) { rk[d] += rk[d + o]; rv[d] += rv[d + o]; }
        __syncthreads();
    }
    kout[(((size_t)b * NH_ + h) * M_ + m) * D_ + d] =
        f2bf(kv * rsqrtf(rk[0] + 1e-12f));
    vtout[(((size_t)b * NH_ + h) * D_ + d) * M_ + m] =
        f2bf(vv * rsqrtf(rv[0] + 1e-12f));
}

// ---------------------------------------------------------------------------
// Flash attention (memory attention): double-buffered K/V tiles filled by
// async global->LDS copies; V supplied pre-transposed so no scatter.
// grid = (S/128, B*NH), block = 256 (8 waves). Wave owns 16 query rows.
// ---------------------------------------------------------------------------
__global__ __launch_bounds__(256)
void attn_kernel(
    const unsigned short* __restrict__ q,   // [b][h][s][d]
    const unsigned short* __restrict__ k,   // [b][h][m][d]
    const unsigned short* __restrict__ vt,  // [b][h][d][m]
    const unsigned char*  __restrict__ mask,// [b][1][s][m]
    unsigned short* __restrict__ ctx) {     // [s*B+b][h*D+d]

    __shared__ __align__(16) unsigned short Ks[2][64 * 136];   // [m][d] pad 8
    __shared__ __align__(16) unsigned short Vt[2][128 * 72];   // [d][m] pad 8
    __shared__ __align__(16) unsigned short Ps[8 * 16 * 72];   // per-wave [r][m]

    const int tid  = threadIdx.x;
    const int w    = tid >> 5;
    const int lane = tid & 31;
    const int half = lane >> 4;
    const int l    = lane & 15;
    const int bh   = blockIdx.y;
    const int b    = bh / NH_;
    const int h    = bh % NH_;
    const int sbase = blockIdx.x * 128 + w * 16;

    const size_t head_off = (size_t)bh * M_ * D_;   // same for q/k/vt
    const int krow0 = tid >> 4;                     // K stage: 0..15 (+16 per t)
    const int kdb   = (tid & 15) * 8;
    const int vrow0 = tid >> 3;                     // V stage: 0..31 (+32 per t)
    const int vmb   = (tid & 7) * 8;

    // load q A-fragments (16 rows x 128 D -> 4 frags of 16x32)
    v16bf qa[4];
#pragma unroll
    for (int kk = 0; kk < 4; ++kk) {
        const unsigned short* qp =
            q + head_off + (size_t)(sbase + l) * D_ + kk * 32 + half * 8;
        FragU tmp;
        tmp.u[0] = *(const uint4*)qp;
        tmp.u[1] = *(const uint4*)(qp + 16);
        qa[kk] = tmp.v;
    }

    v8f acc[8];
#pragma unroll
    for (int i = 0; i < 8; ++i) acc[i] = (v8f)0.0f;
    float mstat[8], lstat[8];
#pragma unroll
    for (int e = 0; e < 8; ++e) { mstat[e] = -3.0e38f; lstat[e] = 0.f; }

    // prologue: async-stage chunk 0 into buffer 0
#pragma unroll
    for (int t = 0; t < 4; ++t) {
        async_copy_b128(&Ks[0][(krow0 + t * 16) * 136 + kdb],
                        k + head_off + (size_t)(krow0 + t * 16) * D_ + kdb);
        async_copy_b128(&Vt[0][(vrow0 + t * 32) * 72 + vmb],
                        vt + head_off + (size_t)(vrow0 + t * 32) * M_ + vmb);
    }

    int buf = 0;
    for (int mc = 0; mc < M_ / 64; ++mc) {
        const int mbase = mc * 64;
        wait_async0();        // copies into current buffer complete
        __syncthreads();      // all threads' copies visible; Ps reads done
        const bool haveNext = (mc + 1 < M_ / 64);
        if (haveNext) {
            int mn = mbase + 64, nb = buf ^ 1;
#pragma unroll
            for (int t = 0; t < 4; ++t) {
                async_copy_b128(&Ks[nb][(krow0 + t * 16) * 136 + kdb],
                                k + head_off +
                                (size_t)(mn + krow0 + t * 16) * D_ + kdb);
                async_copy_b128(&Vt[nb][(vrow0 + t * 32) * 72 + vmb],
                                vt + head_off +
                                (size_t)(vrow0 + t * 32) * M_ + mn + vmb);
            }
        }

        // scores: 4 col-frags of 16x16, K-dim = D accumulated over 4 steps
        v8f sf[4];
#pragma unroll
        for (int nf = 0; nf < 4; ++nf) {
            v8f sacc = (v8f)0.0f;
#pragma unroll
            for (int kk = 0; kk < 4; ++kk) {
                const unsigned short* kp =
                    &Ks[buf][(nf * 16 + l) * 136 + kk * 32 + half * 8];
                FragU tb;
                tb.u[0] = *(const uint4*)kp;
                tb.u[1] = *(const uint4*)(kp + 16);
                sacc = __builtin_amdgcn_wmma_f32_16x16x32_bf16(
                    false, qa[kk], false, tb.v, (short)0, sacc, false, false);
            }
            sf[nf] = sacc;
        }

        // scale + mask
#pragma unroll
        for (int nf = 0; nf < 4; ++nf) {
#pragma unroll
            for (int e = 0; e < 8; ++e) {
                int srow = sbase + e + 8 * half;
                int mcol = mbase + nf * 16 + l;
                float sc = sf[nf][e] * INV_NORM_;
                if (mask[((size_t)b * S_ + srow) * M_ + mcol]) sc = -1e4f;
                sf[nf][e] = sc;
            }
        }

        // online softmax stats (row = across 16 lanes of half-wave)
        float alpha[8];
#pragma unroll
        for (int e = 0; e < 8; ++e) {
            float t = fmaxf(fmaxf(sf[0][e], sf[1][e]), fmaxf(sf[2][e], sf[3][e]));
#pragma unroll
            for (int off = 1; off < 16; off <<= 1)
                t = fmaxf(t, __shfl_xor(t, off, 32));
            float mn = fmaxf(mstat[e], t);
            alpha[e] = __expf(mstat[e] - mn);
            mstat[e] = mn;
        }
#pragma unroll
        for (int nf = 0; nf < 4; ++nf)
#pragma unroll
            for (int e = 0; e < 8; ++e)
                sf[nf][e] = __expf(sf[nf][e] - mstat[e]);
#pragma unroll
        for (int e = 0; e < 8; ++e) {
            float t = sf[0][e] + sf[1][e] + sf[2][e] + sf[3][e];
#pragma unroll
            for (int off = 1; off < 16; off <<= 1)
                t += __shfl_xor(t, off, 32);
            lstat[e] = lstat[e] * alpha[e] + t;
        }
#pragma unroll
        for (int df = 0; df < 8; ++df)
#pragma unroll
            for (int e = 0; e < 8; ++e)
                acc[df][e] *= alpha[e];

        // probs -> LDS (per-wave region), reshape into A-fragments
        unsigned short* pw = Ps + w * (16 * 72);
#pragma unroll
        for (int nf = 0; nf < 4; ++nf)
#pragma unroll
            for (int e = 0; e < 8; ++e)
                pw[(e + 8 * half) * 72 + nf * 16 + l] = f2bf(sf[nf][e]);
        __syncthreads();

        // ctx += P @ V : K-dim = 64 (2 steps), N-dim = 128 (8 frags)
#pragma unroll
        for (int kk = 0; kk < 2; ++kk) {
            const unsigned short* pp = pw + l * 72 + kk * 32 + half * 8;
            FragU pa;
            pa.u[0] = *(const uint4*)pp;
            pa.u[1] = *(const uint4*)(pp + 16);
#pragma unroll
            for (int df = 0; df < 8; ++df) {
                const unsigned short* vp =
                    &Vt[buf][(df * 16 + l) * 72 + kk * 32 + half * 8];
                FragU vb;
                vb.u[0] = *(const uint4*)vp;
                vb.u[1] = *(const uint4*)(vp + 16);
                acc[df] = __builtin_amdgcn_wmma_f32_16x16x32_bf16(
                    false, pa.v, false, vb.v, (short)0, acc[df], false, false);
            }
        }
        buf ^= 1;
    }

    // epilogue: ctx / l, write bf16 in dense-GEMM layout
#pragma unroll
    for (int df = 0; df < 8; ++df) {
#pragma unroll
        for (int e = 0; e < 8; ++e) {
            int srow = sbase + e + 8 * half;
            int d = df * 16 + l;
            float val = acc[df][e] / lstat[e];
            ctx[((size_t)srow * B_ + b) * H_ + h * D_ + d] = f2bf(val);
        }
    }
}

// ---------------------------------------------------------------------------
// Launch
// ---------------------------------------------------------------------------
extern "C" void kernel_launch(void* const* d_in, const int* in_sizes, int n_in,
                              void* d_out, int out_size, void* d_ws, size_t ws_size,
                              hipStream_t stream) {
    const float* x          = (const float*)d_in[0];
    const float* mem_hidden = (const float*)d_in[1];
    const unsigned char* mem_mask = (const unsigned char*)d_in[2];
    const float* ln1_g = (const float*)d_in[3];
    const float* ln1_b = (const float*)d_in[4];
    const float* ln2_g = (const float*)d_in[5];
    const float* ln2_b = (const float*)d_in[6];
    const float* ln3_g = (const float*)d_in[7];
    const float* ln3_b = (const float*)d_in[8];
    const float* w_mlp1_in  = (const float*)d_in[9];
    const float* b_mlp1_in  = (const float*)d_in[10];
    const float* w_mlp1_out = (const float*)d_in[11];
    const float* b_mlp1_out = (const float*)d_in[12];
    const float* w_q   = (const float*)d_in[13];
    const float* b_q   = (const float*)d_in[14];
    const float* w_kv  = (const float*)d_in[15];
    const float* b_kv  = (const float*)d_in[16];
    const float* w_dense = (const float*)d_in[17];
    const float* b_dense = (const float*)d_in[18];
    const float* w_mlp2_in  = (const float*)d_in[19];
    const float* b_mlp2_in  = (const float*)d_in[20];
    const float* w_mlp2_out = (const float*)d_in[21];
    const float* b_mlp2_out = (const float*)d_in[22];
    float* out = (float*)d_out;

    char* ws = (char*)d_ws;
    size_t off = 0;
    auto alloc = [&](size_t bytes) -> void* {
        void* p = ws + off;
        off += (bytes + 255) & ~(size_t)255;
        return p;
    };

    // all weight copies stored TRANSPOSED as bf16 [N][K]
    unsigned short* wb1i  = (unsigned short*)alloc((size_t)H_ * FF_ * 2);
    unsigned short* wb1o  = (unsigned short*)alloc((size_t)FF_ * H_ * 2);
    unsigned short* wb2i  = (unsigned short*)alloc((size_t)H_ * FF_ * 2);
    unsigned short* wb2o  = (unsigned short*)alloc((size_t)FF_ * H_ * 2);
    unsigned short* wbq   = (unsigned short*)alloc((size_t)H_ * H_ * 2);
    unsigned short* wbkv  = (unsigned short*)alloc((size_t)H_ * 2 * H_ * 2);
    unsigned short* wbd   = (unsigned short*)alloc((size_t)H_ * H_ * 2);
    unsigned short* memhb = (unsigned short*)alloc((size_t)MBROWS_ * H_ * 2);
    unsigned short* hbuf  = (unsigned short*)alloc((size_t)ROWS_ * H_ * 2);
    unsigned short* abuf  = (unsigned short*)alloc((size_t)ROWS_ * FF_ * 2);
    float*          xbuf  = (float*)alloc((size_t)ROWS_ * H_ * 4);
    float*          x2buf = (float*)alloc((size_t)ROWS_ * H_ * 4);
    float*          scrf  = (float*)alloc((size_t)ROWS_ * 2 * H_ * 4);
    unsigned short* qb    = (unsigned short*)alloc((size_t)B_ * NH_ * S_ * D_ * 2);
    unsigned short* kb    = (unsigned short*)alloc((size_t)B_ * NH_ * M_ * D_ * 2);
    unsigned short* vtb   = (unsigned short*)alloc((size_t)B_ * NH_ * M_ * D_ * 2);
    unsigned short* ctxb  = (unsigned short*)alloc((size_t)ROWS_ * H_ * 2);
    (void)ws_size; (void)n_in; (void)in_sizes; (void)out_size;

    // weight conversions to transposed bf16
    auto cvtT = [&](const float* src, unsigned short* dst, int K, int N) {
        cvt_transpose_kernel<<<dim3(N / 32, K / 32), 256, 0, stream>>>(src, dst, K, N);
    };
    cvtT(w_mlp1_in,  wb1i, H_, FF_);
    cvtT(w_mlp1_out, wb1o, FF_, H_);
    cvtT(w_mlp2_in,  wb2i, H_, FF_);
    cvtT(w_mlp2_out, wb2o, FF_, H_);
    cvtT(w_q,        wbq,  H_, H_);
    cvtT(w_kv,       wbkv, H_, 2 * H_);
    cvtT(w_dense,    wbd,  H_, H_);
    cvt_bf16_kernel<<<1024, 256, 0, stream>>>(mem_hidden, memhb,
                                              (size_t)MBROWS_ * H_);

    dim3 blk(256);

    // x = MLP1(LN1(x))   (no residual, per reference)
    layernorm_kernel<<<ROWS_, blk, 0, stream>>>(x, ln1_g, ln1_b, hbuf, H_);
    gemm_bf16_kernel<<<dim3(FF_ / 128, ROWS_ / 128), blk, 0, stream>>>(
        hbuf, wb1i, b_mlp1_in, nullptr, nullptr, abuf, ROWS_, FF_, H_, 1);
    gemm_bf16_kernel<<<dim3(H_ / 128, ROWS_ / 128), blk, 0, stream>>>(
        abuf, wb1o, b_mlp1_out, nullptr, xbuf, nullptr, ROWS_, H_, FF_, 0);

    // q = l2norm(LN2(x) @ Wq + bq)
    layernorm_kernel<<<ROWS_, blk, 0, stream>>>(xbuf, ln2_g, ln2_b, hbuf, H_);
    gemm_bf16_kernel<<<dim3(H_ / 128, ROWS_ / 128), blk, 0, stream>>>(
        hbuf, wbq, b_q, nullptr, scrf, nullptr, ROWS_, H_, H_, 0);
    l2norm_q_kernel<<<S_ * B_ * NH_, 128, 0, stream>>>(scrf, qb);

    // k,v = l2norm(mem_hidden @ Wkv + bkv);  v stored transposed per head
    gemm_bf16_kernel<<<dim3(2 * H_ / 128, MBROWS_ / 128), blk, 0, stream>>>(
        memhb, wbkv, b_kv, nullptr, scrf, nullptr, MBROWS_, 2 * H_, H_, 0);
    l2norm_kv_kernel<<<M_ * B_ * NH_, 128, 0, stream>>>(scrf, kb, vtb);

    // ctx = softmax(mask(qk/sqrt(D))) @ v
    attn_kernel<<<dim3(S_ / 128, B_ * NH_), blk, 0, stream>>>(
        qb, kb, vtb, mem_mask, ctxb);

    // x = x + ctx @ Wd + bd
    gemm_bf16_kernel<<<dim3(H_ / 128, ROWS_ / 128), blk, 0, stream>>>(
        ctxb, wbd, b_dense, xbuf, x2buf, nullptr, ROWS_, H_, H_, 0);

    // out = x + MLP2(LN3(x))
    layernorm_kernel<<<ROWS_, blk, 0, stream>>>(x2buf, ln3_g, ln3_b, hbuf, H_);
    gemm_bf16_kernel<<<dim3(FF_ / 128, ROWS_ / 128), blk, 0, stream>>>(
        hbuf, wb2i, b_mlp2_in, nullptr, nullptr, abuf, ROWS_, FF_, H_, 1);
    gemm_bf16_kernel<<<dim3(H_ / 128, ROWS_ / 128), blk, 0, stream>>>(
        abuf, wb2o, b_mlp2_out, x2buf, out, nullptr, ROWS_, H_, FF_, 0);
}
